// merge_attention_69123203661949
// MI455X (gfx1250) — matrix operand order repaired and checked by reading, not compiled, not run on hardware
//
#include <hip/hip_runtime.h>
#include <hip/hip_bf16.h>
#include <math.h>

// ---------------------------------------------------------------------------
// Types for WMMA
// ---------------------------------------------------------------------------
typedef __attribute__((ext_vector_type(16))) _Float16 v16h;
typedef __attribute__((ext_vector_type(8)))  float    v8f;

#define LNN   0   // C = A @ B          (A [M,K] f32 rm, B [K,N] f32 rm)
#define LNT   1   // C = A @ B^T        (A [M,K] f32 rm, B [N,K] f32 rm)
#define LTN16 2   // C = A^T @ B        (A [K,M] f16 rm, B [K,N] f32 rm)

// ---------------------------------------------------------------------------
// Block-tiled WMMA GEMM for gfx1250 (wave32):
//   * block = 256 threads = 8 waves; block C-tile = 128 (M) x 64 (N)
//   * wave w owns rows [blkM*128 + w*16, +16), all 64 cols -> 4 accumulators
//   * K-stage = 64: a 64x64 B tile is cooperatively loaded (coalesced float4,
//     loads batched into a clause before any LDS store), converted to f16 and
//     scattered into LDS in exact WMMA-B lane order; then each wave runs two
//     32-deep substeps: 1 A-tile load + 4 v_wmma_f32_16x16x32_f16 each
//     (8 WMMAs per stage, 2 barriers per stage)
//   * next stage's A segment is prefetched (global_prefetch_b8)
// Per-lane register layouts used here (contiguous k order, same for A and B):
//   A 16x32 f16 : lane<16 -> M=lane, elems 0..15 = K0..15
//                 lane>=16 -> M=lane-16, elems 0..15 = K16..31
//   B 32x16 f16 : lane<16 -> N=lane, elem j = K j ; lane>=16 -> K 16+j
//   C 16x16 f32 : VGPR i : lane<16 -> (M=i, N=lane) ; lane>=16 -> (M=8+i)
// Requires: N % 64 == 0, K % 64 == 0 (all shapes comply); M may be ragged.
// ---------------------------------------------------------------------------
template <int LAYOUT, bool RELU, bool BIAS, bool RESID, bool OUTF16>
__global__ __launch_bounds__(256) void gemm_wmma_k(
    const void* __restrict__ Av, const float* __restrict__ B,
    const float* __restrict__ bias, const float* __restrict__ resid,
    void* __restrict__ Cv, int M, int N, int K,
    int lda, int ldb, int ldc, float alpha)
{
  // 64(K) x 64(N) f16 B tile; substep ks at ldsB[ks*2048 + ct*512 + lane*16 + j]
  __shared__ alignas(32) _Float16 ldsB[4096];

  const int tid   = threadIdx.x;
  const int lane  = tid & 31;
  const int wave  = tid >> 5;
  const int nBlkN = N >> 6;                      // 64-wide column blocks
  const int blkM  = blockIdx.x / nBlkN;
  const int blkN  = blockIdx.x % nBlkN;
  const int row0  = blkM * 128 + wave * 16;
  const int colB  = blkN * 64;
  const int half  = lane >> 4;                   // 0 | 1
  const int l15   = lane & 15;

  v8f acc[4] = {};

  for (int k0 = 0; k0 < K; k0 += 64) {
    __syncthreads();                             // prior reads done before refill

    // ---- phase 1: batched, coalesced global loads (float4) ----
    float4 tmp[4];
#pragma unroll
    for (int i = 0; i < 4; ++i) {
      const int g = i * 256 + tid;               // 1024 float4 groups
      if (LAYOUT == LNT) {                       // B [N,K] rm: contiguous in K
        const int kr = (g & 15) * 4;             // 0..60
        const int n  = g >> 4;                   // 0..63
        tmp[i] = *(const float4*)&B[(size_t)(colB + n) * ldb + k0 + kr];
      } else {                                   // B [K,N] rm: contiguous in N
        const int n  = (g & 15) * 4;             // 0..60
        const int kr = g >> 4;                   // 0..63
        tmp[i] = *(const float4*)&B[(size_t)(k0 + kr) * ldb + colB + n];
      }
    }
    // ---- phase 2: convert f32->f16, scatter to WMMA-B lane order in LDS ----
#pragma unroll
    for (int i = 0; i < 4; ++i) {
      const int g = i * 256 + tid;
      const float* v = &tmp[i].x;
      if (LAYOUT == LNT) {
        const int kr  = (g & 15) * 4;
        const int n   = g >> 4;
        const int ks  = kr >> 5;
        const int krr = kr & 31;
        _Float16* d = &ldsB[ks * 2048 + (n >> 4) * 512 +
                            (((krr >> 4) << 4) + (n & 15)) * 16 + (krr & 15)];
        d[0] = (_Float16)v[0]; d[1] = (_Float16)v[1];      // 4 contiguous halfs
        d[2] = (_Float16)v[2]; d[3] = (_Float16)v[3];      // (ds_store_b64)
      } else {
        const int nb  = (g & 15) * 4;
        const int kr  = g >> 4;
        const int ks  = kr >> 5;
        const int krr = kr & 31;
#pragma unroll
        for (int u = 0; u < 4; ++u) {
          const int n = nb + u;
          ldsB[ks * 2048 + (n >> 4) * 512 +
               (((krr >> 4) << 4) + (n & 15)) * 16 + (krr & 15)] = (_Float16)v[u];
        }
      }
    }
    __syncthreads();

    // ---- two 32-deep WMMA substeps ----
#pragma unroll
    for (int ks = 0; ks < 2; ++ks) {
      const int kk = k0 + ks * 32;
      v16h a = {};
      if (LAYOUT == LNN || LAYOUT == LNT) {
        const float* A = (const float*)Av;
        const int r = row0 + l15;
        if (r < M) {
          const float* p = A + (size_t)r * lda + kk;
          const int kb0 = half * 16;             // 0 | 16
          const int kb1 = half * 16 + 8;         // 8 | 24
#pragma unroll
          for (int j = 0; j < 8; ++j) a[j]     = (_Float16)p[kb0 + j];
#pragma unroll
          for (int j = 0; j < 8; ++j) a[8 + j] = (_Float16)p[kb1 + j];
          if (ks == 0 && k0 + 64 < K)
            __builtin_prefetch(p + 64, 0, 1);    // global_prefetch_b8
        }
      } else {  // LTN16: A is f16 [K,M] rm; lane-coalesced column reads of A^T
        const _Float16* A = (const _Float16*)Av;
        const int r = row0 + l15;
        if (r < M) {
          const int kb0 = half * 16;
          const int kb1 = half * 16 + 8;
#pragma unroll
          for (int j = 0; j < 8; ++j) a[j]     = A[(size_t)(kk + kb0 + j) * lda + r];
#pragma unroll
          for (int j = 0; j < 8; ++j) a[8 + j] = A[(size_t)(kk + kb1 + j) * lda + r];
          if (ks == 0 && k0 + 64 < K)
            __builtin_prefetch(A + (size_t)(k0 + 64) * lda + r, 0, 1);
        }
      }
#pragma unroll
      for (int ct = 0; ct < 4; ++ct) {
        const v16h b = *(const v16h*)(&ldsB[ks * 2048 + ct * 512 + lane * 16]);
        acc[ct] = __builtin_amdgcn_wmma_f32_16x16x32_f16(
            /*neg_a=*/false, a, /*neg_b=*/false, b,
            /*c_mod=*/(short)0, acc[ct], /*reuse_a=*/false, /*reuse_b=*/false);
      }
    }
  }

  // ---- epilogue ----
  const int rbase = row0 + half * 8;
#pragma unroll
  for (int ct = 0; ct < 4; ++ct) {
    const int n = colB + ct * 16 + l15;
    const float bv = BIAS ? bias[n] : 0.0f;
#pragma unroll
    for (int i = 0; i < 8; ++i) {
      const int r = rbase + i;
      if (r < M) {
        float v = acc[ct][i] * alpha + bv;
        if (RELU)  v = fmaxf(v, 0.0f);
        if (RESID) v += resid[(size_t)r * ldc + n];
        if (OUTF16) ((_Float16*)Cv)[(size_t)r * ldc + n] = (_Float16)v;
        else        ((float*)Cv)[(size_t)r * ldc + n]    = v;
      }
    }
  }
}

// ---------------------------------------------------------------------------
// Softmax over each row of the f16 logits (in place) + scores[row] = 1/Z
// (max of a softmax row is exp(0)/Z = 1/Z).
// ---------------------------------------------------------------------------
__global__ __launch_bounds__(256) void softmax_rows_k(_Float16* __restrict__ attn,
                                                      float* __restrict__ scores, int n)
{
  __shared__ float red[256];
  const int row = blockIdx.x;
  const int tid = threadIdx.x;
  _Float16* p = attn + (size_t)row * n;

  float m = -3.0e38f;
  for (int c = tid; c < n; c += 256) m = fmaxf(m, (float)p[c]);
  red[tid] = m; __syncthreads();
  for (int s = 128; s > 0; s >>= 1) {
    if (tid < s) red[tid] = fmaxf(red[tid], red[tid + s]);
    __syncthreads();
  }
  m = red[0]; __syncthreads();

  float z = 0.0f;
  for (int c = tid; c < n; c += 256) z += expf((float)p[c] - m);
  red[tid] = z; __syncthreads();
  for (int s = 128; s > 0; s >>= 1) {
    if (tid < s) red[tid] += red[tid + s];
    __syncthreads();
  }
  z = red[0];
  const float inv = 1.0f / z;
  for (int c = tid; c < n; c += 256) p[c] = (_Float16)(expf((float)p[c] - m) * inv);
  if (tid == 0) scores[row] = inv;
}

// ---------------------------------------------------------------------------
// Single-workgroup bitonic argsort of 8192 (score,idx) pairs in LDS.
// Strict total order (score desc, idx asc) == stable argsort(-scores).
// ---------------------------------------------------------------------------
__global__ __launch_bounds__(1024) void bitonic_argsort_k(const float* __restrict__ scores,
                                                          int* __restrict__ idx_out)
{
  __shared__ float s[8192];
  __shared__ int   id[8192];
  const int tid = threadIdx.x;
  for (int i = tid; i < 8192; i += 1024) { s[i] = scores[i]; id[i] = i; }
  __syncthreads();
  for (int k = 2; k <= 8192; k <<= 1) {
    for (int j = k >> 1; j > 0; j >>= 1) {
      for (int i = tid; i < 8192; i += 1024) {
        const int ixj = i ^ j;
        if (ixj > i) {
          const float sa = s[i], sb = s[ixj];
          const int   ia = id[i], ib = id[ixj];
          const bool aBeforeB = (sa > sb) || (sa == sb && ia < ib);
          const bool desc = ((i & k) == 0);
          if (desc ? !aBeforeB : aBeforeB) {
            s[i] = sb; s[ixj] = sa; id[i] = ib; id[ixj] = ia;
          }
        }
      }
      __syncthreads();
    }
  }
  for (int i = tid; i < 8192; i += 1024) idx_out[i] = id[i];
}

// sortedx[i] = (attn^T x)[i] + x[idx[i]]   (the only place the sort matters)
__global__ __launch_bounds__(256) void add_gather_x_k(const float* __restrict__ gATx,
                                                      const float* __restrict__ x,
                                                      const int* __restrict__ idx,
                                                      float* __restrict__ sortedx, int d)
{
  const int i   = blockIdx.x;
  const int src = idx[i];
  for (int c = threadIdx.x; c < d; c += 256)
    sortedx[(size_t)i * d + c] = gATx[(size_t)i * d + c] + x[(size_t)src * d + c];
}

// merged[m][c] = h2[2m][c] + h2[2m+1][c]   (reshape(-1,2,64).sum(axis=1))
__global__ __launch_bounds__(256) void pair_merge_k(const float* __restrict__ h2,
                                                    float* __restrict__ merged, int total, int c64)
{
  const int t = blockIdx.x * 256 + threadIdx.x;
  if (t < total) {
    const int m = t / c64, c = t % c64;
    merged[t] = h2[(size_t)(2 * m) * c64 + c] + h2[(size_t)(2 * m + 1) * c64 + c];
  }
}

// low[c] = sum_{i=r0..r1-1} sortedx[i][c]
__global__ __launch_bounds__(256) void low_sum_k(const float* __restrict__ sortedx,
                                                 float* __restrict__ low, int d, int r0, int r1)
{
  const int c = blockIdx.x * 256 + threadIdx.x;
  if (c < d) {
    float acc = 0.0f;
    for (int i = r0; i < r1; ++i) acc += sortedx[(size_t)i * d + c];
    low[c] = acc;
  }
}

__global__ __launch_bounds__(256) void layernorm_k(const float* __restrict__ in,
                                                   const float* __restrict__ gamma,
                                                   const float* __restrict__ beta,
                                                   float* __restrict__ out, int d, float eps)
{
  __shared__ float red[256];
  const int r   = blockIdx.x;
  const int tid = threadIdx.x;
  const float* p = in + (size_t)r * d;

  float s = 0.0f;
  for (int c = tid; c < d; c += 256) s += p[c];
  red[tid] = s; __syncthreads();
  for (int q = 128; q > 0; q >>= 1) { if (tid < q) red[tid] += red[tid + q]; __syncthreads(); }
  const float mu = red[0] / (float)d; __syncthreads();

  float v = 0.0f;
  for (int c = tid; c < d; c += 256) { const float t = p[c] - mu; v += t * t; }
  red[tid] = v; __syncthreads();
  for (int q = 128; q > 0; q >>= 1) { if (tid < q) red[tid] += red[tid + q]; __syncthreads(); }
  const float rstd = rsqrtf(red[0] / (float)d + eps);

  for (int c = tid; c < d; c += 256)
    out[(size_t)r * d + c] = (p[c] - mu) * rstd * gamma[c] + beta[c];
}

// ---------------------------------------------------------------------------
// Host side
// ---------------------------------------------------------------------------
static inline int gemm_grid(int M, int N) {
  return ((M + 127) / 128) * (N / 64);   // 128x64 block tiles
}

extern "C" void kernel_launch(void* const* d_in, const int* in_sizes, int n_in,
                              void* d_out, int out_size, void* d_ws, size_t ws_size,
                              hipStream_t stream)
{
  (void)in_sizes; (void)n_in; (void)out_size; (void)ws_size;
  const int N = 8192, D = 512;
  const int HIGH = 4096, MERGED = 2048, OUTR = 2049;  // merge_factor = 2

  const float* x    = (const float*)d_in[0];
  const float* qw1  = (const float*)d_in[1];
  const float* qb1  = (const float*)d_in[2];
  const float* qw2  = (const float*)d_in[3];
  const float* qb2  = (const float*)d_in[4];
  const float* kw1  = (const float*)d_in[5];
  const float* kb1  = (const float*)d_in[6];
  const float* kw2  = (const float*)d_in[7];
  const float* kb2  = (const float*)d_in[8];
  const float* rw1  = (const float*)d_in[9];
  const float* rb1  = (const float*)d_in[10];
  const float* rw2  = (const float*)d_in[11];
  const float* rb2  = (const float*)d_in[12];
  const float* hw   = (const float*)d_in[13];
  const float* hb   = (const float*)d_in[14];
  const float* ow1  = (const float*)d_in[15];
  const float* ob1  = (const float*)d_in[16];
  const float* ow2  = (const float*)d_in[17];
  const float* ob2  = (const float*)d_in[18];
  const float* gam  = (const float*)d_in[19];
  const float* bet  = (const float*)d_in[20];

  // ---- workspace carve (bytes) ----
  char* base = (char*)d_ws;
  size_t off = 0;
  auto carve = [&](size_t bytes) { void* p = base + off; off += (bytes + 255) & ~(size_t)255; return p; };
  _Float16* attn  = (_Float16*)carve((size_t)N * N * sizeof(_Float16)); // 128 MB, fits L2
  float* h1       = (float*)carve((size_t)N * 256 * sizeof(float));
  float* q        = (float*)carve((size_t)N * 128 * sizeof(float));
  float* k        = (float*)carve((size_t)N * 128 * sizeof(float));
  float* scores   = (float*)carve((size_t)N * sizeof(float));
  int*   sidx     = (int*)carve((size_t)N * sizeof(int));
  float* gATx     = (float*)carve((size_t)N * D * sizeof(float));
  float* sortedx  = (float*)carve((size_t)N * D * sizeof(float));
  float* h2a      = (float*)carve((size_t)HIGH * 128 * sizeof(float));
  float* h2b      = (float*)carve((size_t)HIGH * 64 * sizeof(float));
  float* merged   = (float*)carve((size_t)MERGED * 64 * sizeof(float));
  float* out0     = (float*)carve((size_t)OUTR * D * sizeof(float));
  float* o1       = (float*)carve((size_t)OUTR * 128 * sizeof(float));
  float* outpre   = (float*)carve((size_t)OUTR * D * sizeof(float));

  const float inv_sqrt_d = 0.044194173824159216f; // 1/sqrt(512)

  // q path: h1 = relu(x@qw1+qb1); q = h1@qw2+qb2
  gemm_wmma_k<LNN, true,  true,  false, false><<<gemm_grid(N, 256), 256, 0, stream>>>(
      x, qw1, qb1, nullptr, h1, N, 256, D, D, 256, 256, 1.0f);
  gemm_wmma_k<LNN, false, true,  false, false><<<gemm_grid(N, 128), 256, 0, stream>>>(
      h1, qw2, qb2, nullptr, q, N, 128, 256, 256, 128, 128, 1.0f);
  // k path (reuses h1)
  gemm_wmma_k<LNN, true,  true,  false, false><<<gemm_grid(N, 256), 256, 0, stream>>>(
      x, kw1, kb1, nullptr, h1, N, 256, D, D, 256, 256, 1.0f);
  gemm_wmma_k<LNN, false, true,  false, false><<<gemm_grid(N, 128), 256, 0, stream>>>(
      h1, kw2, kb2, nullptr, k, N, 128, 256, 256, 128, 128, 1.0f);

  // logits = (q @ k^T) / sqrt(512), stored f16 (fits in 192MB L2)
  gemm_wmma_k<LNT, false, false, false, true><<<gemm_grid(N, N), 256, 0, stream>>>(
      q, k, nullptr, nullptr, attn, N, N, 128, 128, 128, N, inv_sqrt_d);

  // softmax rows in place (f16), scores[row] = max prob = 1/Z
  softmax_rows_k<<<N, 256, 0, stream>>>(attn, scores, N);

  // stable descending argsort of scores
  bitonic_argsort_k<<<1, 1024, 0, stream>>>(scores, sidx);

  // (P attn)^T (P x) == attn^T x  -> permutation only needed for residual
  gemm_wmma_k<LTN16, false, false, false, false><<<gemm_grid(N, D), 256, 0, stream>>>(
      attn, x, nullptr, nullptr, gATx, N, D, N, N, D, D, 1.0f);
  add_gather_x_k<<<N, 256, 0, stream>>>(gATx, x, sidx, sortedx, D);

  // high path
  gemm_wmma_k<LNN, true,  true,  false, false><<<gemm_grid(HIGH, 128), 256, 0, stream>>>(
      sortedx, rw1, rb1, nullptr, h2a, HIGH, 128, D, D, 128, 128, 1.0f);
  gemm_wmma_k<LNN, false, true,  false, false><<<gemm_grid(HIGH, 64), 256, 0, stream>>>(
      h2a, rw2, rb2, nullptr, h2b, HIGH, 64, 128, 128, 64, 64, 1.0f);
  pair_merge_k<<<(MERGED * 64 + 255) / 256, 256, 0, stream>>>(h2b, merged, MERGED * 64, 64);
  gemm_wmma_k<LNN, false, true,  false, false><<<gemm_grid(MERGED, D), 256, 0, stream>>>(
      merged, hw, hb, nullptr, out0, MERGED, D, 64, 64, D, D, 1.0f);

  // low path -> last row of out0
  low_sum_k<<<(D + 255) / 256, 256, 0, stream>>>(sortedx, out0 + (size_t)MERGED * D, D, HIGH, N);

  // out MLP with residual, then layernorm
  gemm_wmma_k<LNN, true,  true,  false, false><<<gemm_grid(OUTR, 128), 256, 0, stream>>>(
      out0, ow1, ob1, nullptr, o1, OUTR, 128, D, D, 128, 128, 1.0f);
  gemm_wmma_k<LNN, false, true,  true,  false><<<gemm_grid(OUTR, D), 256, 0, stream>>>(
      o1, ow2, ob2, out0, outpre, OUTR, D, 128, 128, D, D, 1.0f);

  layernorm_k<<<OUTR, 256, 0, stream>>>(outpre, gam, bet, (float*)d_out, D, 1e-5f);
}
